// MultiHeadAttention_7894149890677
// MI455X (gfx1250) — matrix operand order repaired and checked by reading, not compiled, not run on hardware
//
#include <hip/hip_runtime.h>

#define BATCH  4
#define SEQ    2048
#define DMODEL 1024
#define NHEAD  16
#define HDIM   64
#define MROWS  (BATCH * SEQ)   // 8192

typedef __bf16 bf16_t;
typedef __attribute__((ext_vector_type(8)))  __bf16 v8bf;
typedef __attribute__((ext_vector_type(16))) __bf16 v16bf;
typedef __attribute__((ext_vector_type(8)))  float  v8f;

#define V8F_ZERO {0.f,0.f,0.f,0.f,0.f,0.f,0.f,0.f}

#define WMMA_BF16(a, b, c) \
  __builtin_amdgcn_wmma_f32_16x16x32_bf16(false, (a), false, (b), (short)0, (c), false, false)

__device__ __forceinline__ v16bf bf_combine(v8bf lo, v8bf hi) {
  return __builtin_shufflevector(lo, hi, 0,1,2,3,4,5,6,7,8,9,10,11,12,13,14,15);
}

// A fragment: 16x32 bf16 matrix, row-major with leading dim ld (elements).
// CDNA5 layout: lane L (L<16) row=L, K = {0..7, 16..23}; lane L>=16 row=L-16, K = {8..15, 24..31}.
__device__ __forceinline__ v16bf load_a_frag(const bf16_t* base, int ld, int lane) {
  const int half = lane >> 4;
  const int row  = lane & 15;
  const bf16_t* p = base + row * ld + half * 8;
  v8bf lo = *reinterpret_cast<const v8bf*>(p);
  v8bf hi = *reinterpret_cast<const v8bf*>(p + 16);
  return bf_combine(lo, hi);
}

// B fragment: 32x16 bf16 (KxN). Source holds B^T row-major: srcT[n][k] == B[k][n].
// CDNA5 layout: lane L (L<16) col=L, K = {0..7, 16..23}; lane L>=16 col=L-16, K = {8..15, 24..31}.
__device__ __forceinline__ v16bf load_b_frag(const bf16_t* baseT, int ld, int lane) {
  const int half = lane >> 4;
  const int col  = lane & 15;
  const bf16_t* p = baseT + col * ld + half * 8;
  v8bf lo = *reinterpret_cast<const v8bf*>(p);
  v8bf hi = *reinterpret_cast<const v8bf*>(p + 16);
  return bf_combine(lo, hi);
}

// ---------------------------------------------------------------------------
// Y = X @ W^T. Block tile 128x64, 4 waves, each wave owns 32x64 (8 WMMA / K-step).
// Ping-pong double-buffered LDS: one barrier per K-step; global loads for the
// next tile are issued before the WMMAs so the matrix pipe hides their latency.
// A_IS_F32: A is f32 (inputs) else bf16 (attention output).
// HEADSPLIT: write bf16 [B,H,N,HD]; else write f32 [MROWS, DMODEL].
// ---------------------------------------------------------------------------
template <bool A_IS_F32, bool HEADSPLIT>
__global__ __launch_bounds__(128)
void gemm_xwT_kernel(const float* __restrict__ Af, const bf16_t* __restrict__ Ab,
                     const float* __restrict__ W, bf16_t* __restrict__ Yb,
                     float* __restrict__ Yf) {
  __shared__ alignas(16) bf16_t As[2][128][32];
  __shared__ alignas(16) bf16_t Ws[2][64][32];   // Ws[n][k] = W[n0+n][k0+k]

  const int t    = threadIdx.x;
  const int wave = t >> 5;
  const int lane = t & 31;
  const int n0   = blockIdx.x * 64;
  const int m0   = blockIdx.y * 128;

  const v8f vzero = V8F_ZERO;
  v8f acc[2][4];
#pragma unroll
  for (int s = 0; s < 2; ++s)
#pragma unroll
    for (int g = 0; g < 4; ++g) acc[s][g] = vzero;

  // Staging registers (one A row of 32 elems per thread; half a W row per thread).
  float4 ra[8];
  int4   rab[4];
  float4 rw[4];
  const int wrow = t >> 1;
  const int wcol = (t & 1) * 16;

  auto fetch = [&](int k0) {
    if (A_IS_F32) {
      const float4* ga = reinterpret_cast<const float4*>(Af + (size_t)(m0 + t) * DMODEL + k0);
#pragma unroll
      for (int i = 0; i < 8; ++i) ra[i] = ga[i];
    } else {
      const int4* ga = reinterpret_cast<const int4*>(Ab + (size_t)(m0 + t) * DMODEL + k0);
#pragma unroll
      for (int i = 0; i < 4; ++i) rab[i] = ga[i];
    }
    const float4* gw = reinterpret_cast<const float4*>(W + (size_t)(n0 + wrow) * DMODEL + k0 + wcol);
#pragma unroll
    for (int i = 0; i < 4; ++i) rw[i] = gw[i];
  };

  auto store_lds = [&](int buf) {
    if (A_IS_F32) {
#pragma unroll
      for (int i = 0; i < 8; ++i) {
        As[buf][t][4*i + 0] = (bf16_t)ra[i].x; As[buf][t][4*i + 1] = (bf16_t)ra[i].y;
        As[buf][t][4*i + 2] = (bf16_t)ra[i].z; As[buf][t][4*i + 3] = (bf16_t)ra[i].w;
      }
    } else {
#pragma unroll
      for (int i = 0; i < 4; ++i)
        *reinterpret_cast<int4*>(&As[buf][t][8 * i]) = rab[i];
    }
#pragma unroll
    for (int i = 0; i < 4; ++i) {
      Ws[buf][wrow][wcol + 4*i + 0] = (bf16_t)rw[i].x; Ws[buf][wrow][wcol + 4*i + 1] = (bf16_t)rw[i].y;
      Ws[buf][wrow][wcol + 4*i + 2] = (bf16_t)rw[i].z; Ws[buf][wrow][wcol + 4*i + 3] = (bf16_t)rw[i].w;
    }
  };

  fetch(0);
  store_lds(0);
  __syncthreads();

  const int NK = DMODEL / 32;
  for (int kt = 0; kt < NK; ++kt) {
    const int cur = kt & 1;
    const bool has_next = (kt + 1 < NK);
    if (has_next) fetch((kt + 1) * 32);          // issue global loads early

    v16bf a0 = load_a_frag(&As[cur][wave * 32][0],      32, lane);
    v16bf a1 = load_a_frag(&As[cur][wave * 32 + 16][0], 32, lane);
#pragma unroll
    for (int g = 0; g < 4; ++g) {
      v16bf b = load_b_frag(&Ws[cur][g * 16][0], 32, lane);
      acc[0][g] = WMMA_BF16(a0, b, acc[0][g]);
      acc[1][g] = WMMA_BF16(a1, b, acc[1][g]);
    }

    if (has_next) store_lds(cur ^ 1);            // other buffer: no read/write conflict
    __syncthreads();
  }

  const int half = lane >> 4;
  const int cl   = lane & 15;
#pragma unroll
  for (int s = 0; s < 2; ++s) {
#pragma unroll
    for (int g = 0; g < 4; ++g) {
      const int col = n0 + g * 16 + cl;
#pragma unroll
      for (int r = 0; r < 8; ++r) {
        const int row = m0 + wave * 32 + s * 16 + half * 8 + r;
        if (HEADSPLIT) {
          const int h = col >> 6, hd = col & 63;
          const int bidx = row >> 11, n = row & (SEQ - 1);
          Yb[(((size_t)bidx * NHEAD + h) * SEQ + n) * HDIM + hd] = (bf16_t)acc[s][g][r];
        } else {
          Yf[(size_t)row * DMODEL + col] = acc[s][g][r];
        }
      }
    }
  }
}

// ---------------------------------------------------------------------------
// Causal flash attention per (head, 64-query tile). 4 waves x 16 q-rows.
// K tile staged via CDNA5 async global->LDS DMA (ASYNCcnt); V transposed manually.
// ---------------------------------------------------------------------------
__global__ __launch_bounds__(128)
void flash_attn_kernel(const bf16_t* __restrict__ Qh, const bf16_t* __restrict__ Kh,
                       const bf16_t* __restrict__ Vh, bf16_t* __restrict__ O) {
  __shared__ alignas(16) bf16_t Kt[32][64];       // Kt[k][d]
  __shared__ alignas(16) bf16_t Vt[64][32];       // Vt[d][k]  (V transposed)
  __shared__ alignas(16) bf16_t Ps[4][16][32];    // per-wave P tile (A-matrix source)

  const int t    = threadIdx.x;
  const int wave = t >> 5;
  const int lane = t & 31;
  const int q0   = blockIdx.x * 64;
  const int bh   = blockIdx.y;
  const int b    = bh >> 4;
  const int h    = bh & 15;

  const bf16_t* Qhead = Qh + (size_t)bh * SEQ * HDIM;
  const bf16_t* Khead = Kh + (size_t)bh * SEQ * HDIM;
  const bf16_t* Vhead = Vh + (size_t)bh * SEQ * HDIM;

  const int qw   = q0 + wave * 16;
  const int half = lane >> 4;
  const int l16  = lane & 15;

  v16bf aQ0 = load_a_frag(Qhead + (size_t)qw * HDIM,      HDIM, lane);
  v16bf aQ1 = load_a_frag(Qhead + (size_t)qw * HDIM + 32, HDIM, lane);

  const float NEGBIG = -1.0e30f;
  float mrow[8], lrow[8];
  const v8f vzero = V8F_ZERO;
  v8f acc[4] = { vzero, vzero, vzero, vzero };
#pragma unroll
  for (int r = 0; r < 8; ++r) { mrow[r] = NEGBIG; lrow[r] = 0.f; }

  const int ntiles = q0 / 32 + 2;       // block-uniform causal bound
  const int srow   = t >> 2;            // 0..31 (key row within tile)
  const int sd0    = (t & 3) * 16;      // 0,16,32,48 (d chunk)

  // LDS byte offset for this thread's K-tile slice (ISA 10.2: flat addr[31:0] == LDS offset).
  const unsigned ldsK = (unsigned)(uintptr_t)&Kt[srow][sd0];

  for (int kt = 0; kt < ntiles; ++kt) {
    const int kb = kt * 32;

    // K tile: async DMA global -> LDS (32 bytes per thread, two b128 ops; the
    // instruction offset applies to both the LDS and global address).
    {
      const bf16_t* gk = Khead + (size_t)(kb + srow) * HDIM + sd0;
      asm volatile("global_load_async_to_lds_b128 %0, %1, off"
                   :: "v"(ldsK), "v"(gk) : "memory");
      asm volatile("global_load_async_to_lds_b128 %0, %1, off offset:16"
                   :: "v"(ldsK), "v"(gk) : "memory");
    }
    // V tile: load row-major, scatter transposed into LDS.
    {
      const v8bf* gv = reinterpret_cast<const v8bf*>(Vhead + (size_t)(kb + srow) * HDIM + sd0);
      v8bf va = gv[0], vb = gv[1];
#pragma unroll
      for (int i = 0; i < 8; ++i) {
        Vt[sd0 + i][srow]     = va[i];
        Vt[sd0 + 8 + i][srow] = vb[i];
      }
    }
    if (kt + 1 < ntiles) {   // prefetch next tile (global_prefetch_b8)
      __builtin_prefetch(Khead + (size_t)(kb + 32 + srow) * HDIM + sd0, 0, 1);
      __builtin_prefetch(Vhead + (size_t)(kb + 32 + srow) * HDIM + sd0, 0, 1);
    }
    asm volatile("s_wait_asynccnt 0" ::: "memory");
    __syncthreads();

    // S[16x32] = Q[16x64] . K^T : two 16x16 accumulators, K-dim split 32+32.
    v8f S0 = vzero, S1 = vzero;
    S0 = WMMA_BF16(aQ0, load_b_frag(&Kt[0][0],   64, lane), S0);
    S0 = WMMA_BF16(aQ1, load_b_frag(&Kt[0][32],  64, lane), S0);
    S1 = WMMA_BF16(aQ0, load_b_frag(&Kt[16][0],  64, lane), S1);
    S1 = WMMA_BF16(aQ1, load_b_frag(&Kt[16][32], 64, lane), S1);

    // Online softmax. C-layout: element r <-> row half*8+r, col l16 (S0) / l16+16 (S1).
    float p0[8], p1[8];
#pragma unroll
    for (int r = 0; r < 8; ++r) {
      const int qg = qw + half * 8 + r;
      const int c0 = kb + l16;
      const int c1 = c0 + 16;
      float s0 = (c0 <= qg) ? S0[r] * 0.125f : NEGBIG;   // 1/sqrt(64)
      float s1 = (c1 <= qg) ? S1[r] * 0.125f : NEGBIG;
      float tm = fmaxf(s0, s1);
#pragma unroll
      for (int off = 1; off < 16; off <<= 1)
        tm = fmaxf(tm, __shfl_xor(tm, off, 32));
      const float mnew  = fmaxf(mrow[r], tm);
      const float scale = __expf(mrow[r] - mnew);
      const float e0 = __expf(s0 - mnew);
      const float e1 = __expf(s1 - mnew);
      float rs = e0 + e1;
#pragma unroll
      for (int off = 1; off < 16; off <<= 1)
        rs += __shfl_xor(rs, off, 32);
      lrow[r] = lrow[r] * scale + rs;
      mrow[r] = mnew;
#pragma unroll
      for (int g = 0; g < 4; ++g) acc[g][r] = acc[g][r] * scale;
      p0[r] = e0;
      p1[r] = e1;
    }

    // Re-materialize P into LDS in A-matrix source layout (wave-private; DS in-order).
#pragma unroll
    for (int r = 0; r < 8; ++r) {
      Ps[wave][half * 8 + r][l16]      = (bf16_t)p0[r];
      Ps[wave][half * 8 + r][l16 + 16] = (bf16_t)p1[r];
    }

    // O[16x64] += P[16x32] . V[32x64]
    v16bf aP = load_a_frag(&Ps[wave][0][0], 32, lane);
#pragma unroll
    for (int g = 0; g < 4; ++g) {
      v16bf bV = load_b_frag(&Vt[g * 16][0], 32, lane);
      acc[g] = WMMA_BF16(aP, bV, acc[g]);
    }
    __syncthreads();
  }

  // Epilogue: normalize and store bf16 to [B, N, H*HD].
#pragma unroll
  for (int g = 0; g < 4; ++g) {
    const int col = h * HDIM + g * 16 + l16;
#pragma unroll
    for (int r = 0; r < 8; ++r) {
      const int qg = qw + half * 8 + r;
      O[((size_t)(b * SEQ + qg)) * DMODEL + col] = (bf16_t)(acc[g][r] / lrow[r]);
    }
  }
}

// ---------------------------------------------------------------------------
extern "C" void kernel_launch(void* const* d_in, const int* in_sizes, int n_in,
                              void* d_out, int out_size, void* d_ws, size_t ws_size,
                              hipStream_t stream) {
  const float* q  = (const float*)d_in[0];
  const float* k  = (const float*)d_in[1];
  const float* v  = (const float*)d_in[2];
  const float* Wq = (const float*)d_in[3];
  const float* Wk = (const float*)d_in[4];
  const float* Wv = (const float*)d_in[5];
  const float* Wo = (const float*)d_in[6];
  float* out = (float*)d_out;

  const size_t elems = (size_t)MROWS * DMODEL;   // 8M elements per buffer
  bf16_t* Qh = (bf16_t*)d_ws;                    // [B,H,N,HD] bf16
  bf16_t* Kh = Qh + elems;
  bf16_t* Vh = Kh + elems;
  bf16_t* AO = Vh + elems;                       // [B*N, D] bf16

  dim3 gGemm(DMODEL / 64, MROWS / 128);          // (16, 64)
  gemm_xwT_kernel<true,  true ><<<gGemm, 128, 0, stream>>>(q, nullptr, Wq, Qh, nullptr);
  gemm_xwT_kernel<true,  true ><<<gGemm, 128, 0, stream>>>(k, nullptr, Wk, Kh, nullptr);
  gemm_xwT_kernel<true,  true ><<<gGemm, 128, 0, stream>>>(v, nullptr, Wv, Vh, nullptr);

  dim3 gAttn(SEQ / 64, BATCH * NHEAD);           // (32, 64)
  flash_attn_kernel<<<gAttn, 128, 0, stream>>>(Qh, Kh, Vh, AO);

  gemm_xwT_kernel<false, false><<<gGemm, 128, 0, stream>>>(nullptr, AO, Wo, nullptr, out);
}
